// EMAVectorQuantizer_66279935311937
// MI455X (gfx1250) — matrix operand order repaired and checked by reading, not compiled, not run on hardware
//
#include <hip/hip_runtime.h>
#include <hip/hip_bf16.h>

typedef __attribute__((ext_vector_type(16))) _Float16 v16h;
typedef __attribute__((ext_vector_type(4)))  _Float16 v4h;
typedef __attribute__((ext_vector_type(8)))  float    v8f;

#define TPB      32            // tokens per block
#define DDIM     1024
#define HEADS    4
#define KCODES   64
#define DH       256           // per-head dim
#define EMB_ELTS (HEADS * KCODES * DH)   // 65536
#define WS_EMB_OFF 32          // byte offset of f16 codebook in d_ws (32B aligned)

__device__ __forceinline__ float wave_sum(float v) {
  #pragma unroll
  for (int off = 16; off > 0; off >>= 1) v += __shfl_xor(v, off, 32);
  return v;
}

// ~6 VALU tanh (1 transcendental v_exp_f32 + fast divide); plenty for argmax/clip.
__device__ __forceinline__ float fast_tanh(float x) {
  const float a = fabsf(x);
  const float e = __expf(-2.0f * a);
  const float t = __fdividef(1.0f - e, 1.0f + e);
  return copysignf(t, x);
}

__global__ void vq_init_kernel(unsigned* __restrict__ ws) {
  if (threadIdx.x < 4) ws[threadIdx.x] = 0u;   // [0..1] presence mask, [2] loss bits
}

// one-time fp32 -> f16 codebook conversion into workspace
__global__ void vq_cvt_emb_kernel(const float* __restrict__ emb,
                                  _Float16* __restrict__ embh) {
  const int i = (blockIdx.x * 256 + threadIdx.x) * 4;
  const float4 v = *(const float4*)(emb + i);
  v4h h;
  h.x = (_Float16)v.x; h.y = (_Float16)v.y;
  h.z = (_Float16)v.z; h.w = (_Float16)v.w;
  *(v4h*)(embh + i) = h;
}

template <bool F16EMB>
__global__ __launch_bounds__(256) void vq_fused_kernel(
    const float* __restrict__ inp, const float* __restrict__ lnw,
    const float* __restrict__ lnb, const float* __restrict__ emb,
    const _Float16* __restrict__ embh,
    float* __restrict__ out, unsigned* __restrict__ ws)
{
  __shared__ _Float16 xh[TPB * DDIM];              // 64 KB: xn in f16
  __shared__ float    dist[TPB * HEADS * KCODES];  // 32 KB: dot products
  __shared__ int      sidx[TPB][HEADS];
  __shared__ unsigned pres[2];
  __shared__ float    wsum[8];

  const int t    = threadIdx.x;
  const int lane = t & 31;
  const int w    = t >> 5;                 // wave id, 8 waves
  const int tb   = blockIdx.x * TPB;

  if (t < 2) pres[t] = 0u;

  // ---------------- Phase 1: LayerNorm + tanh clip + L2 normalize ----------
  // wave w handles tokens 4w..4w+3; 32 lanes cover 1024 elems (32 each).
  #pragma unroll
  for (int r = 0; r < 4; ++r) {
    const int tk = w * 4 + r;
    const float* ip = inp + (size_t)(tb + tk) * DDIM;
    float4 v[8];
    float s = 0.f, s2 = 0.f;
    #pragma unroll
    for (int j = 0; j < 8; ++j) {
      v[j] = *(const float4*)(ip + j * 128 + lane * 4);
      s  += v[j].x + v[j].y + v[j].z + v[j].w;
      s2 += v[j].x * v[j].x + v[j].y * v[j].y + v[j].z * v[j].z + v[j].w * v[j].w;
    }
    s  = wave_sum(s);
    s2 = wave_sum(s2);
    const float mean = s * (1.0f / DDIM);
    const float var  = s2 * (1.0f / DDIM) - mean * mean;
    const float rstd = rsqrtf(var + 1e-5f);

    float ss = 0.f;
    #pragma unroll
    for (int j = 0; j < 8; ++j) {
      const float4 wv = *(const float4*)(lnw + j * 128 + lane * 4);
      const float4 bv = *(const float4*)(lnb + j * 128 + lane * 4);
      float4 x;
      x.x = 5.0f * fast_tanh(0.2f * ((v[j].x - mean) * rstd * wv.x + bv.x));
      x.y = 5.0f * fast_tanh(0.2f * ((v[j].y - mean) * rstd * wv.y + bv.y));
      x.z = 5.0f * fast_tanh(0.2f * ((v[j].z - mean) * rstd * wv.z + bv.z));
      x.w = 5.0f * fast_tanh(0.2f * ((v[j].w - mean) * rstd * wv.w + bv.w));
      v[j] = x;
      ss += x.x * x.x + x.y * x.y + x.z * x.z + x.w * x.w;
    }
    ss = wave_sum(ss);
    const float invn = 1.0f / fmaxf(sqrtf(ss), 1e-5f);

    _Float16* xr = xh + tk * DDIM;
    #pragma unroll
    for (int j = 0; j < 8; ++j) {
      const int base = j * 128 + lane * 4;
      v4h h;
      h.x = (_Float16)(v[j].x * invn);
      h.y = (_Float16)(v[j].y * invn);
      h.z = (_Float16)(v[j].z * invn);
      h.w = (_Float16)(v[j].w * invn);
      *(v4h*)(xr + base) = h;
    }
  }
  __syncthreads();

  // ---------------- Phase 2: WMMA dist = xn(16xK) . emb^T(Kx16) ------------
  // 32 output tiles (2 m-tiles x 4 heads x 4 n-tiles); wave w does tiles w+8r.
  const int m  = lane & 15;
  const int hi = lane >> 4;
  #pragma unroll
  for (int r = 0; r < 4; ++r) {
    const int tl   = w + 8 * r;          // 0..31
    const int mt   = tl >> 4;
    const int head = (tl >> 2) & 3;
    const int nt   = tl & 3;
    const _Float16* arow = xh + (mt * 16 + m) * DDIM + head * DH;
    const int ak = hi * 16;   // A-fragment K base (contiguous: k = 16*hi + s, matches B)
    const int bk = hi * 16;   // B-fragment K base (lane->N, halves->K)
    v8f acc = {};
    for (int kk = 0; kk < 8; ++kk) {
      const int k0 = kk * 32;
      v16h a, b;
      #pragma unroll
      for (int i = 0; i < 8; ++i) {
        a[i]     = arow[k0 + ak + i];
        a[i + 8] = arow[k0 + ak + 8 + i];
      }
      if (F16EMB) {
        // 32B-aligned contiguous 16 halves straight from the f16 codebook
        b = *(const v16h*)(embh + (size_t)(head * KCODES + nt * 16 + m) * DH + k0 + bk);
      } else {
        const float* bp = emb + (size_t)(head * KCODES + nt * 16 + m) * DH + k0 + bk;
        #pragma unroll
        for (int i = 0; i < 16; ++i) b[i] = (_Float16)bp[i];
      }
      acc = __builtin_amdgcn_wmma_f32_16x16x32_f16(
          /*neg_a=*/false, a, /*neg_b=*/false, b,
          /*c_mod=*/(short)0, acc, /*reuse_a=*/false, /*reuse_b=*/false);
    }
    #pragma unroll
    for (int vv = 0; vv < 8; ++vv)   // C/D layout: M = vv + 8*hi, N = m
      dist[(mt * 16 + vv + hi * 8) * (HEADS * KCODES) + head * KCODES + nt * 16 + m] = acc[vv];
  }
  __syncthreads();

  // ---------------- Phase 3: argmax over 64 codes per (token, head) --------
  if (t < TPB * HEADS) {
    const int tk = t >> 2;
    const int hd = t & 3;
    const float* dr = dist + tk * (HEADS * KCODES) + hd * KCODES;
    float best = dr[0];
    int bi = 0;
    for (int k = 1; k < KCODES; ++k) {
      float dv = dr[k];
      if (dv > best) { best = dv; bi = k; }   // first-max tie break, like argmax
    }
    sidx[tk][hd] = bi;
    atomicOr(&pres[bi >> 5], 1u << (bi & 31));
  }
  __syncthreads();
  if (t < 2) atomicOr(&ws[t], pres[t]);

  // ---------------- Phase 4: gather q, write out, accumulate loss ----------
  {
    const int l16 = t & 15;
    float lsum = 0.f;
    #pragma unroll
    for (int rr = 0; rr < 2; ++rr) {
      const int tk = (t >> 4) + 16 * rr;
      float* orow = out + (size_t)(tb + tk) * DDIM;
      const _Float16* xrow = xh + tk * DDIM;
      #pragma unroll 4
      for (int e = 0; e < 64; ++e) {
        const int d  = e * 16 + l16;
        const int hd = d >> 8;
        const int k  = sidx[tk][hd];
        const float qv = emb[(size_t)(hd * KCODES + k) * DH + (d & 255)];
        orow[d] = qv;                       // quantized == q in forward value
        const float diff = qv - (float)xrow[d];
        lsum += diff * diff;
      }
    }
    lsum = wave_sum(lsum);
    if (lane == 0) wsum[w] = lsum;
  }
  __syncthreads();
  if (t == 0) {
    float tot = 0.f;
    #pragma unroll
    for (int i = 0; i < 8; ++i) tot += wsum[i];
    atomicAdd((float*)&ws[2], tot);
  }
}

__global__ void vq_final_kernel(const unsigned* __restrict__ ws,
                                float* __restrict__ out,
                                unsigned long long qcount) {
  if (threadIdx.x == 0 && blockIdx.x == 0) {
    const int cnt = __popc(ws[0]) + __popc(ws[1]);
    const float sumsq = ((const float*)ws)[2];
    out[qcount]     = sumsq * (0.25f / (float)qcount);  // COMMITMENT_COST * mean
    out[qcount + 1] = (float)cnt;                        // unique_count
  }
}

extern "C" void kernel_launch(void* const* d_in, const int* in_sizes, int n_in,
                              void* d_out, int out_size, void* d_ws, size_t ws_size,
                              hipStream_t stream) {
  (void)in_sizes; (void)n_in; (void)out_size;
  const float* inp = (const float*)d_in[0];  // (8,4096,1024)
  const float* lnw = (const float*)d_in[1];  // (1024,)
  const float* lnb = (const float*)d_in[2];  // (1024,)
  const float* emb = (const float*)d_in[3];  // (4,64,256)
  float* out   = (float*)d_out;
  unsigned* ws = (unsigned*)d_ws;
  _Float16* embh = (_Float16*)((char*)d_ws + WS_EMB_OFF);

  const int ntok = 8 * 4096;                 // 32768 tokens
  const size_t ws_need = WS_EMB_OFF + (size_t)EMB_ELTS * sizeof(_Float16);

  vq_init_kernel<<<1, 64, 0, stream>>>(ws);
  if (ws_size >= ws_need) {
    vq_cvt_emb_kernel<<<EMB_ELTS / (256 * 4), 256, 0, stream>>>(emb, embh);
    vq_fused_kernel<true><<<ntok / TPB, 256, 0, stream>>>(inp, lnw, lnb, emb, embh, out, ws);
  } else {
    vq_fused_kernel<false><<<ntok / TPB, 256, 0, stream>>>(inp, lnw, lnb, emb, embh, out, ws);
  }
  vq_final_kernel<<<1, 64, 0, stream>>>(ws, out, (unsigned long long)ntok * DDIM);
}
